// NodeEmbDecoder_88716844466371
// MI455X (gfx1250) — compile-verified, matrix operand order checked by reading, not run on hardware
//
#include <hip/hip_runtime.h>
#include <hip/hip_bf16.h>

// Problem constants (from reference): B=256, E=512, D=128, H=512, L=2, N=128
#define BB 256
#define EE 512
#define DD 128
#define HH 512
#define NN 128

#define NBLK  64
#define NTHR  256
#define NWAVE (NBLK * 8)   // 512 wave32 waves

typedef __bf16 bf16;
typedef __attribute__((ext_vector_type(16))) __bf16 v16bf;
typedef __attribute__((ext_vector_type(8)))  __bf16 v8bf;
typedef __attribute__((ext_vector_type(8)))  float  v8f;

// ---------------------------------------------------------------------------
// WMMA 16x16x32 bf16 fragment loader.
// A (16x32, row-major [M,K]) per CDNA5 layout: lanes 0-15 hold rows M=lane with
// K = {k0..k0+7, k0+16..k0+23}; lanes 16-31 hold K = {k0+8..k0+15, k0+24..k0+31}.
// B (32x16) mirrors A with N<->M, so a row-major [N,K] weight matrix loads with
// the same addressing (K contiguous per output column).
// ---------------------------------------------------------------------------
__device__ __forceinline__ v16bf frag(const bf16* __restrict__ m, int ld,
                                      int row0, int k0, int lane) {
  const bf16* p = m + (size_t)(row0 + (lane & 15)) * ld + (k0 + ((lane >> 4) << 3));
  v8bf lo = *(const v8bf*)(p);
  v8bf hi = *(const v8bf*)(p + 16);
  return __builtin_shufflevector(lo, hi, 0,1,2,3,4,5,6,7,8,9,10,11,12,13,14,15);
}

#define WMMA_BF16(A, B, C) \
  __builtin_amdgcn_wmma_f32_16x16x32_bf16(false, (A), false, (B), (short)0, (C), false, false)

__device__ __forceinline__ float sigm(float x) { return 1.0f / (1.0f + __expf(-x)); }
__device__ __forceinline__ float tanh_(float x) {
  float e = __expf(-2.0f * x);
  return (1.0f - e) / (1.0f + e);
}

// Sense-reversing device-wide barrier (persistent kernel, 64 resident blocks).
__device__ void gbar(unsigned* bar) {
  __syncthreads();
  if (threadIdx.x == 0) {
    volatile unsigned* gen = bar + 1;
    unsigned g = *gen;
    __threadfence();
    if (atomicAdd(bar, 1u) == (unsigned)(NBLK - 1)) {
      bar[0] = 0;
      __threadfence();
      atomicAdd((unsigned*)(bar + 1), 1u);
    } else {
      while (*gen == g) { __builtin_amdgcn_s_sleep(1); }
      __threadfence();
    }
  }
  __syncthreads();
}

// ---------------------------------------------------------------------------
// Setup kernels
// ---------------------------------------------------------------------------
__global__ void k_zero(uint4* p, int n16) {
  int i = blockIdx.x * blockDim.x + threadIdx.x;
  if (i < n16) { uint4 z; z.x = z.y = z.z = z.w = 0u; p[i] = z; }
}

__global__ void k_cvt(const float* __restrict__ s, bf16* __restrict__ d, int n) {
  int i = blockIdx.x * blockDim.x + threadIdx.x;
  if (i < n) d[i] = (bf16)s[i];
}

__global__ void k_nodesq(const float* __restrict__ node, float* __restrict__ sq) {
  int i = blockIdx.x * blockDim.x + threadIdx.x;  // over B*N
  if (i >= BB * NN) return;
  const float* p = node + (size_t)i * DD;
  float s = 0.f;
  #pragma unroll 4
  for (int d = 0; d < DD; ++d) s += p[d] * p[d];
  sq[i] = s;
}

// ---------------------------------------------------------------------------
// Persistent decoder: FNN init + 128 sequential steps of
//   2-layer LSTM -> projection -> masked argmin -> scatter
// ---------------------------------------------------------------------------
__global__ __launch_bounds__(NTHR, 1)
void decoder_kernel(
    const float* __restrict__ node,
    const float* __restrict__ b1v, const float* __restrict__ b2v,
    const float* __restrict__ bih0, const float* __restrict__ bhh0,
    const float* __restrict__ bih1, const float* __restrict__ bhh1,
    const float* __restrict__ bov, const int* __restrict__ tfp,
    const bf16* __restrict__ w1, const bf16* __restrict__ w2,
    const bf16* __restrict__ wih0, const bf16* __restrict__ whh0,
    const bf16* __restrict__ wih1, const bf16* __restrict__ whh1,
    const bf16* __restrict__ wo, const bf16* __restrict__ embb,
    bf16* h0a, bf16* h0b, bf16* h1a, bf16* h1b,
    bf16* tmp, float* pred, const float* __restrict__ nodesq,
    float* c0, float* c1, bf16* decb, int* used,
    unsigned* bar, float* out)
{
  const int lane  = threadIdx.x & 31;
  const int wv    = threadIdx.x >> 5;
  const int gw    = blockIdx.x * 8 + wv;     // global wave id 0..511
  const int rbase = (lane >> 4) << 3;        // C/D: lanes 16-31 hold rows M+8
  const int cl    = lane & 15;
  __shared__ float spred[8][DD];

  // ---- FNN stage 1: tmp = relu(emb @ W1^T + b1)  [256,512], 512 tiles
  {
    int m = gw >> 5, n = gw & 31;
    v8f acc = {};
    for (int k = 0; k < EE / 32; ++k) {
      v16bf a = frag(embb, EE, m * 16, k * 32, lane);
      v16bf b = frag(w1,   EE, n * 16, k * 32, lane);
      acc = WMMA_BF16(a, b, acc);
    }
    int col = n * 16 + cl;
    float bb = b1v[col];
    #pragma unroll
    for (int r = 0; r < 8; ++r) {
      int row = m * 16 + rbase + r;
      float v = acc[r] + bb;
      tmp[(size_t)row * HH + col] = (bf16)(v > 0.f ? v : 0.f);
    }
  }
  gbar(bar);

  // ---- FNN stage 2: [h0|h1] = tmp @ W2^T + b2  [256,1024], 1024 tiles
  for (int t = gw; t < 16 * 64; t += NWAVE) {
    int m = t >> 6, n = t & 63;
    v8f acc = {};
    for (int k = 0; k < HH / 32; ++k) {
      v16bf a = frag(tmp, HH, m * 16, k * 32, lane);
      v16bf b = frag(w2,  HH, n * 16, k * 32, lane);
      acc = WMMA_BF16(a, b, acc);
    }
    int col = n * 16 + cl;
    float bb = b2v[col];
    bf16* dst = (col < HH) ? h0a : h1a;
    int cc = col & (HH - 1);
    #pragma unroll
    for (int r = 0; r < 8; ++r) {
      int row = m * 16 + rbase + r;
      dst[(size_t)row * HH + cc] = (bf16)(acc[r] + bb);
    }
  }
  gbar(bar);

  const int tf = *tfp;

  for (int s = 0; s < NN; ++s) {
    const bf16* h0p = (s & 1) ? h0b : h0a;
    bf16*       h0n = (s & 1) ? h0a : h0b;
    const bf16* h1p = (s & 1) ? h1b : h1a;
    bf16*       h1n = (s & 1) ? h1a : h1b;

    // ---- LSTM layer 0: gates = dec@Wih0^T + h0@Whh0^T; fuse cell in regs.
    // Wave owns (batch tile m, hidden tile j) and all 4 gate tiles for it.
    {
      int m = gw >> 5, j = gw & 31;
      v8f acc[4] = { {}, {}, {}, {} };
      for (int k = 0; k < DD / 32; ++k) {
        v16bf a = frag(decb, DD, m * 16, k * 32, lane);
        #pragma unroll
        for (int q = 0; q < 4; ++q) {
          v16bf b = frag(wih0, DD, q * HH + j * 16, k * 32, lane);
          acc[q] = WMMA_BF16(a, b, acc[q]);
        }
      }
      for (int k = 0; k < HH / 32; ++k) {
        v16bf a = frag(h0p, HH, m * 16, k * 32, lane);
        #pragma unroll
        for (int q = 0; q < 4; ++q) {
          v16bf b = frag(whh0, HH, q * HH + j * 16, k * 32, lane);
          acc[q] = WMMA_BF16(a, b, acc[q]);
        }
      }
      int col = j * 16 + cl;
      float bi  = bih0[col]          + bhh0[col];
      float bf_ = bih0[HH + col]     + bhh0[HH + col];
      float bg  = bih0[2 * HH + col] + bhh0[2 * HH + col];
      float bo_ = bih0[3 * HH + col] + bhh0[3 * HH + col];
      #pragma unroll
      for (int r = 0; r < 8; ++r) {
        int row = m * 16 + rbase + r;
        size_t o = (size_t)row * HH + col;
        float cn = sigm(acc[1][r] + bf_) * c0[o] +
                   sigm(acc[0][r] + bi)  * tanh_(acc[2][r] + bg);
        c0[o] = cn;
        h0n[o] = (bf16)(sigm(acc[3][r] + bo_) * tanh_(cn));
      }
    }
    gbar(bar);

    // ---- LSTM layer 1: input = h0n
    {
      int m = gw >> 5, j = gw & 31;
      v8f acc[4] = { {}, {}, {}, {} };
      for (int k = 0; k < HH / 32; ++k) {
        v16bf a = frag(h0n, HH, m * 16, k * 32, lane);
        #pragma unroll
        for (int q = 0; q < 4; ++q) {
          v16bf b = frag(wih1, HH, q * HH + j * 16, k * 32, lane);
          acc[q] = WMMA_BF16(a, b, acc[q]);
        }
      }
      for (int k = 0; k < HH / 32; ++k) {
        v16bf a = frag(h1p, HH, m * 16, k * 32, lane);
        #pragma unroll
        for (int q = 0; q < 4; ++q) {
          v16bf b = frag(whh1, HH, q * HH + j * 16, k * 32, lane);
          acc[q] = WMMA_BF16(a, b, acc[q]);
        }
      }
      int col = j * 16 + cl;
      float bi  = bih1[col]          + bhh1[col];
      float bf_ = bih1[HH + col]     + bhh1[HH + col];
      float bg  = bih1[2 * HH + col] + bhh1[2 * HH + col];
      float bo_ = bih1[3 * HH + col] + bhh1[3 * HH + col];
      #pragma unroll
      for (int r = 0; r < 8; ++r) {
        int row = m * 16 + rbase + r;
        size_t o = (size_t)row * HH + col;
        float cn = sigm(acc[1][r] + bf_) * c1[o] +
                   sigm(acc[0][r] + bi)  * tanh_(acc[2][r] + bg);
        c1[o] = cn;
        h1n[o] = (bf16)(sigm(acc[3][r] + bo_) * tanh_(cn));
      }
    }
    gbar(bar);

    // ---- projection: pred = h1n @ Wo^T + bo  [256,128], 128 tiles (blocks 0..15)
    if (blockIdx.x < 16) {
      int m = gw >> 3, n = gw & 7;
      v8f acc = {};
      for (int k = 0; k < HH / 32; ++k) {
        v16bf a = frag(h1n, HH, m * 16, k * 32, lane);
        v16bf b = frag(wo,  HH, n * 16, k * 32, lane);
        acc = WMMA_BF16(a, b, acc);
      }
      int col = n * 16 + cl;
      float bb = bov[col];
      #pragma unroll
      for (int r = 0; r < 8; ++r) {
        int row = m * 16 + rbase + r;
        float v = acc[r] + bb;
        pred[(size_t)row * DD + col] = v;
        decb[(size_t)row * DD + col] = (bf16)v;   // teacher_forcing == 0 path
      }
    }
    gbar(bar);

    // ---- masked argmin + permutation scatter (blocks 0..31, one wave per b)
    if (blockIdx.x < 32) {
      int b = gw;                                 // 0..255
      float* sp = spred[wv];                      // wave-private LDS row
      for (int d = lane; d < DD; d += 32) sp[d] = pred[(size_t)b * DD + d];
      // argmin_n of ||pred - node||^2  == argmin of (||node||^2 - 2 pred.node)
      float best = 3.0e38f; int bestn = 0;
      #pragma unroll
      for (int t4 = 0; t4 < 4; ++t4) {
        int n = lane + t4 * 32;
        float v;
        if (used[b * NN + n]) {
          v = 3.0e38f;
        } else {
          const float* nd = node + ((size_t)b * NN + n) * DD;
          float dot = 0.f;
          #pragma unroll 4
          for (int d = 0; d < DD; ++d) dot += sp[d] * nd[d];
          v = nodesq[b * NN + n] - 2.0f * dot;
        }
        if (v < best || (v == best && n < bestn)) { best = v; bestn = n; }
      }
      #pragma unroll
      for (int off = 16; off; off >>= 1) {
        float ov = __shfl_xor(best, off, 32);
        int   on = __shfl_xor(bestn, off, 32);
        if (ov < best || (ov == best && on < bestn)) { best = ov; bestn = on; }
      }
      if (lane == 0) used[b * NN + bestn] = 1;
      float* orow = out + ((size_t)b * NN + bestn) * DD;
      const float* nrow = node + ((size_t)b * NN + bestn) * DD;
      for (int d = lane; d < DD; d += 32) {
        orow[d] = sp[d];
        if (tf) decb[(size_t)b * DD + d] = (bf16)nrow[d];
      }
    }
    gbar(bar);
  }
}

// ---------------------------------------------------------------------------
extern "C" void kernel_launch(void* const* d_in, const int* in_sizes, int n_in,
                              void* d_out, int out_size, void* d_ws, size_t ws_size,
                              hipStream_t stream) {
  const float* emb  = (const float*)d_in[0];
  const float* node = (const float*)d_in[1];
  const float* W1   = (const float*)d_in[2];
  const float* b1   = (const float*)d_in[3];
  const float* W2   = (const float*)d_in[4];
  const float* b2   = (const float*)d_in[5];
  const float* Wih0 = (const float*)d_in[6];
  const float* Whh0 = (const float*)d_in[7];
  const float* bih0 = (const float*)d_in[8];
  const float* bhh0 = (const float*)d_in[9];
  const float* Wih1 = (const float*)d_in[10];
  const float* Whh1 = (const float*)d_in[11];
  const float* bih1 = (const float*)d_in[12];
  const float* bhh1 = (const float*)d_in[13];
  const float* Wo   = (const float*)d_in[14];
  const float* bo   = (const float*)d_in[15];
  const int*   tf   = (const int*)d_in[16];
  float* out = (float*)d_out;

  char* ws = (char*)d_ws;
  size_t off = 0;
  auto alloc = [&](size_t bytes) -> char* {
    char* p = ws + off;
    off += (bytes + 255) & ~(size_t)255;
    return p;
  };

  // --- zero-initialized state (contiguous so one k_zero covers it) ---
  unsigned* bar  = (unsigned*)alloc(256);
  float*    c0   = (float*)alloc((size_t)BB * HH * 4);
  float*    c1   = (float*)alloc((size_t)BB * HH * 4);
  bf16*     decb = (bf16*) alloc((size_t)BB * DD * 2);
  int*      used = (int*)  alloc((size_t)BB * NN * 4);
  size_t zero_bytes = off;                    // 256-aligned => /16 exact

  // --- persistent activations / converted weights ---
  bf16*  h0a    = (bf16*) alloc((size_t)BB * HH * 2);
  bf16*  h0b    = (bf16*) alloc((size_t)BB * HH * 2);
  bf16*  h1a    = (bf16*) alloc((size_t)BB * HH * 2);
  bf16*  h1b    = (bf16*) alloc((size_t)BB * HH * 2);
  bf16*  tmp    = (bf16*) alloc((size_t)BB * HH * 2);
  float* pred   = (float*)alloc((size_t)BB * DD * 4);
  float* nodesq = (float*)alloc((size_t)BB * NN * 4);
  bf16*  embb   = (bf16*) alloc((size_t)BB * EE * 2);
  bf16*  w1b    = (bf16*) alloc((size_t)HH * EE * 2);
  bf16*  w2b    = (bf16*) alloc((size_t)2 * HH * HH * 2);
  bf16*  wih0b  = (bf16*) alloc((size_t)4 * HH * DD * 2);
  bf16*  whh0b  = (bf16*) alloc((size_t)4 * HH * HH * 2);
  bf16*  wih1b  = (bf16*) alloc((size_t)4 * HH * HH * 2);
  bf16*  whh1b  = (bf16*) alloc((size_t)4 * HH * HH * 2);
  bf16*  wob    = (bf16*) alloc((size_t)DD * HH * 2);

  // --- setup ---
  int zn = (int)(zero_bytes / 16);
  k_zero<<<(zn + 255) / 256, 256, 0, stream>>>((uint4*)ws, zn);

  auto cvt = [&](const float* s, bf16* d, int n) {
    k_cvt<<<(n + 255) / 256, 256, 0, stream>>>(s, d, n);
  };
  cvt(emb,  embb,  BB * EE);
  cvt(W1,   w1b,   HH * EE);
  cvt(W2,   w2b,   2 * HH * HH);
  cvt(Wih0, wih0b, 4 * HH * DD);
  cvt(Whh0, whh0b, 4 * HH * HH);
  cvt(Wih1, wih1b, 4 * HH * HH);
  cvt(Whh1, whh1b, 4 * HH * HH);
  cvt(Wo,   wob,   DD * HH);

  k_nodesq<<<(BB * NN + 255) / 256, 256, 0, stream>>>(node, nodesq);

  // --- persistent decode ---
  decoder_kernel<<<NBLK, NTHR, 0, stream>>>(
      node, b1, b2, bih0, bhh0, bih1, bhh1, bo, tf,
      w1b, w2b, wih0b, whh0b, wih1b, whh1b, wob, embb,
      h0a, h0b, h1a, h1b, tmp, pred, nodesq,
      c0, c1, decb, used, bar, out);
}
